// ScaledCosineAttention_87651692577374
// MI455X (gfx1250) — compile-verified, matrix operand order checked by reading, not run on hardware
//
#include <hip/hip_runtime.h>
#include <hip/hip_bf16.h>

// ---------------------------------------------------------------------------
// Scaled cosine attention: B=4, N=2048, C=768, H=12, D=64
//   q = l2norm(x@Wq+bq) * D^-0.5 ; k,v from x@Wkv+bkv
//   out = softmax(q k^T) v @ Wproj + bproj
// bf16 WMMA (v_wmma_f32_16x16x32_bf16) for all GEMMs; flash-style online
// softmax; Tensor Data Mover (tensor_load_to_lds + s_wait_tensorcnt) for the
// contiguous Q/K tile copies in the attention kernel.
// ---------------------------------------------------------------------------

#define Bb 4
#define Nn 2048
#define Cc 768
#define Hh 12
#define Dd 64
#define BN (Bb * Nn)       // 8192
#define NC3 (3 * Cc)       // 2304

using bf16 = __bf16;
typedef __attribute__((ext_vector_type(16))) __bf16 bf16x16;
typedef __attribute__((ext_vector_type(8)))  __bf16 bf16x8;
typedef __attribute__((ext_vector_type(8)))  float  floatx8;
typedef __attribute__((ext_vector_type(4)))  unsigned int v4u;
typedef __attribute__((ext_vector_type(4)))  int v4i;
typedef __attribute__((ext_vector_type(8)))  int v8i;

union ABFrag { bf16x16 v; bf16x8 h[2]; };

static __device__ __forceinline__ bf16 f2bf(float f) { return (bf16)f; }       // v_cvt bf16 (RTNE)
static __device__ __forceinline__ float bf2f(bf16 b) { return (float)b; }

#define WMMA_BF16(A, Bm, Cacc) \
    __builtin_amdgcn_wmma_f32_16x16x32_bf16(false, (A), false, (Bm), (short)0, (Cacc), false, false)

// ---------------------------------------------------------------------------
// Tensor Data Mover: 1-D contiguous copy of n_elems bf16 (2B) global -> LDS.
// D# per CDNA5 ISA ch.8: group0 = {count=1, lds_addr, global_addr, type=2},
// group1 = {data_size=2B, tensor_dim0=tile_dim0=n_elems, tensor_dim1=1}.
// Issue from ONE wave (TDM ignores EXEC); follow with s_wait_tensorcnt.
// ---------------------------------------------------------------------------
static __device__ __forceinline__ void tdm_copy_2b(const void* gsrc, void* lds_dst,
                                                   unsigned n_elems)
{
    unsigned long long ga = (unsigned long long)(uintptr_t)gsrc;
    v4u g0;
    g0[0] = 1u;                                           // count=1, user D#
    g0[1] = (unsigned)(uintptr_t)lds_dst;                 // lds_addr
    g0[2] = (unsigned)ga;                                 // global_addr[31:0]
    g0[3] = (unsigned)((ga >> 32) & 0x01FFFFFFu) | (2u << 30); // addr[56:32] | type=2
    v8i g1;
    g1[0] = 0x00010000;                                   // data_size = 2 bytes
    g1[1] = (int)((n_elems & 0xFFFFu) << 16);             // tensor_dim0[15:0]
    g1[2] = (int)((n_elems >> 16) & 0xFFFFu) | (1 << 16); // tensor_dim0[31:16] | tensor_dim1=1
    g1[3] = (int)((n_elems & 0xFFFFu) << 16);             // tile_dim0
    g1[4] = 0;                                            // tile_dim1/2 unused
    g1[5] = (int)n_elems;                                 // tensor_dim0_stride lo
    g1[6] = 0;
    g1[7] = 0;
    v4i z4 = {0, 0, 0, 0};
#if defined(__clang_major__) && (__clang_major__ >= 23)
    v8i z8 = {0, 0, 0, 0, 0, 0, 0, 0};
    __builtin_amdgcn_tensor_load_to_lds(g0, g1, z4, z4, z8, 0);
#else
    __builtin_amdgcn_tensor_load_to_lds(g0, g1, z4, z4, 0);
#endif
}

// ---------------------------------------------------------------------------
// Kernel 1: fused QKV projection.  X[8192x768] @ [Wq|Wkv][768x2304] + bias.
// 64x64 tiles; 256 threads = 8 waves, each wave owns a 16x32 sub-tile.
// Writes Q,K,V as bf16 in [B,H,N,D] layout.
// ---------------------------------------------------------------------------
__global__ __launch_bounds__(256) void qkv_gemm_kernel(
    const float* __restrict__ X, const float* __restrict__ Wq,
    const float* __restrict__ bq, const float* __restrict__ Wkv,
    const float* __restrict__ bkv, bf16* __restrict__ qo,
    bf16* __restrict__ ko, bf16* __restrict__ vo)
{
    __shared__ bf16 lA[64][32];   // [m][k]
    __shared__ bf16 lB[64][32];   // [n][k]  (weight tile, transposed)

    const int tid  = threadIdx.x;
    const int lane = tid & 31;
    const int wave = tid >> 5;
    const int m0   = blockIdx.y * 64;
    const int n0   = blockIdx.x * 64;
    const int wm   = (wave >> 1) * 16;
    const int wn   = (wave & 1) * 32;
    const int l15  = lane & 15;
    const int kh   = (lane >> 4) * 8;

    floatx8 acc[2];
#pragma unroll
    for (int t = 0; t < 2; ++t)
#pragma unroll
        for (int i = 0; i < 8; ++i) acc[t][i] = 0.0f;

    for (int kt = 0; kt < Cc / 32; ++kt) {
        const int k0 = kt * 32;
        // stage A: 64x32 f32 -> bf16, one 16B LDS store per thread
        {
            const int r  = tid >> 2;
            const int c0 = (tid & 3) * 8;
            const float* src = X + (size_t)(m0 + r) * Cc + k0 + c0;
            if (kt + 1 < Cc / 32) __builtin_prefetch(src + 32, 0, 1);  // global_prefetch_b8
            bf16x8 pk;
#pragma unroll
            for (int i = 0; i < 8; ++i) pk[i] = f2bf(src[i]);
            *reinterpret_cast<bf16x8*>(&lA[r][c0]) = pk;
        }
        // stage B transposed: lB[n][k]
        {
            const int n  = tid & 63;
            const int kq = (tid >> 6) * 8;
            const int j  = n0 + n;
            const float* wsrc;
            int col, ld;
            if (j < Cc) { wsrc = Wq;  col = j;      ld = Cc;     }
            else        { wsrc = Wkv; col = j - Cc; ld = 2 * Cc; }
            bf16x8 pk;
#pragma unroll
            for (int i = 0; i < 8; ++i)
                pk[i] = f2bf(wsrc[(size_t)(k0 + kq + i) * ld + col]);
            *reinterpret_cast<bf16x8*>(&lB[n][kq]) = pk;
        }
        __syncthreads();

        ABFrag a;
        a.h[0] = *reinterpret_cast<const bf16x8*>(&lA[wm + l15][kh]);
        a.h[1] = *reinterpret_cast<const bf16x8*>(&lA[wm + l15][16 + kh]);
#pragma unroll
        for (int t = 0; t < 2; ++t) {
            ABFrag bb;
            const int n = wn + t * 16 + l15;
            bb.h[0] = *reinterpret_cast<const bf16x8*>(&lB[n][kh]);
            bb.h[1] = *reinterpret_cast<const bf16x8*>(&lB[n][16 + kh]);
            acc[t] = WMMA_BF16(a.v, bb.v, acc[t]);
        }
        __syncthreads();
    }

    // epilogue: bias + scatter to [B,H,N,D] bf16
    const int hi8 = (lane >> 4) << 3;   // C/D layout: lanes 16-31 hold rows r+8
#pragma unroll
    for (int t = 0; t < 2; ++t) {
        const int j = n0 + wn + t * 16 + l15;
        float bias; bf16* dst; int jj;
        if (j < Cc)          { bias = bq[j];       dst = qo; jj = j;          }
        else if (j < 2 * Cc) { bias = bkv[j - Cc]; dst = ko; jj = j - Cc;     }
        else                 { bias = bkv[j - Cc]; dst = vo; jj = j - 2 * Cc; }
        const int h = jj >> 6, d = jj & 63;
#pragma unroll
        for (int r = 0; r < 8; ++r) {
            const int gi = m0 + wm + hi8 + r;
            const int b  = gi >> 11;
            const int n  = gi & (Nn - 1);
            dst[(((size_t)(b * Hh + h)) * Nn + n) * Dd + d] = f2bf(acc[t][r] + bias);
        }
    }
}

// ---------------------------------------------------------------------------
// Kernel 2: per-row (D=64) L2 normalize in place; optionally fold in scale.
// ---------------------------------------------------------------------------
__global__ __launch_bounds__(256) void l2norm_kernel(
    bf16* __restrict__ t, const float* __restrict__ scale, int applyScale)
{
    const int lane = threadIdx.x & 31;
    const int wave = threadIdx.x >> 5;
    const size_t row = (size_t)blockIdx.x * 8 + wave;   // row in [0, B*H*N)
    bf16* p = t + row * Dd;

    float x0 = bf2f(p[lane]);
    float x1 = bf2f(p[lane + 32]);
    float s = x0 * x0 + x1 * x1;
#pragma unroll
    for (int m = 1; m <= 16; m <<= 1) s += __shfl_xor(s, m, 32);
    float r = 1.0f / fmaxf(sqrtf(s), 1e-12f);
    if (applyScale) {
        const int h = (int)((row / Nn) % Hh);
        r *= scale[h];
    }
    p[lane]      = f2bf(x0 * r);
    p[lane + 32] = f2bf(x1 * r);
}

// ---------------------------------------------------------------------------
// Kernel 3: flash attention.  grid = (N/64, B*H), 128 threads = 4 waves.
// Q tile and K chunks arrive via the Tensor Data Mover; V is transposed on
// the LDS-store path (B-fragments of P@V need column-major over seq).
// ---------------------------------------------------------------------------
__global__ __launch_bounds__(128) void flash_attn_kernel(
    const bf16* __restrict__ Q, const bf16* __restrict__ K,
    const bf16* __restrict__ V, bf16* __restrict__ O)
{
    __shared__ bf16 lQ[64][64];
    __shared__ bf16 lK[64][64];    // [n][d] — rows are B-fragments of K^T
    __shared__ bf16 lVt[64][64];   // [d][n] — transposed V chunk
    __shared__ bf16 lP[64][64];    // probabilities, per-wave 16-row bands

    const int tid  = threadIdx.x;
    const int lane = tid & 31;
    const int wave = tid >> 5;
    const int l15  = lane & 15;
    const int kh   = (lane >> 4) * 8;
    const int hi8  = (lane >> 4) << 3;
    const int bh   = blockIdx.y;        // b*H + h
    const int qt   = blockIdx.x;        // 64-query tile

    // async Q tile load (one TDM op; entire 64x64 tile is contiguous)
    const size_t qbase = ((size_t)bh * Nn + qt * 64) * Dd;
    if (wave == 0) tdm_copy_2b(Q + qbase, &lQ[0][0], 64 * 64);

    floatx8 Oacc[4];
#pragma unroll
    for (int t = 0; t < 4; ++t)
#pragma unroll
        for (int i = 0; i < 8; ++i) Oacc[t][i] = 0.0f;
    float mx[8], ls[8];
#pragma unroll
    for (int r = 0; r < 8; ++r) { mx[r] = -1e30f; ls[r] = 0.0f; }

    for (int c0 = 0; c0 < Nn; c0 += 64) {
        __syncthreads();   // previous iteration done reading lK/lVt
        const size_t baseK = ((size_t)bh * Nn + c0) * Dd;
        if (wave == 0) tdm_copy_2b(K + baseK, &lK[0][0], 64 * 64);   // async K chunk
        // V chunk: vector loads + transposed LDS scatter (overlaps the TDM copy)
        for (int i = tid; i < 512; i += 128) {
            const int r = i >> 3, c = (i & 7) * 8;
            bf16x8 vv = *reinterpret_cast<const bf16x8*>(V + baseK + (size_t)r * Dd + c);
#pragma unroll
            for (int j = 0; j < 8; ++j) lVt[c + j][r] = vv[j];
        }
        if (wave == 0) __builtin_amdgcn_s_wait_tensorcnt((short)0);
        __syncthreads();

        // S = Q K^T (16x64 per wave)
        floatx8 s[4];
#pragma unroll
        for (int t = 0; t < 4; ++t)
#pragma unroll
            for (int i = 0; i < 8; ++i) s[t][i] = 0.0f;

        const int qrow = wave * 16 + l15;
#pragma unroll
        for (int kk = 0; kk < 64; kk += 32) {
            ABFrag a;
            a.h[0] = *reinterpret_cast<const bf16x8*>(&lQ[qrow][kk + kh]);
            a.h[1] = *reinterpret_cast<const bf16x8*>(&lQ[qrow][kk + 16 + kh]);
#pragma unroll
            for (int t = 0; t < 4; ++t) {
                ABFrag bb;
                const int kn = t * 16 + l15;
                bb.h[0] = *reinterpret_cast<const bf16x8*>(&lK[kn][kk + kh]);
                bb.h[1] = *reinterpret_cast<const bf16x8*>(&lK[kn][kk + 16 + kh]);
                s[t] = WMMA_BF16(a.v, bb.v, s[t]);
            }
        }

        // online softmax over each of the wave's 16 rows
#pragma unroll
        for (int r = 0; r < 8; ++r) {
            float rm = s[0][r];
#pragma unroll
            for (int t = 1; t < 4; ++t) rm = fmaxf(rm, s[t][r]);
#pragma unroll
            for (int m = 1; m <= 8; m <<= 1) rm = fmaxf(rm, __shfl_xor(rm, m, 32));
            const float mnew  = fmaxf(mx[r], rm);
            const float alpha = __expf(mx[r] - mnew);
            float rs = 0.0f;
#pragma unroll
            for (int t = 0; t < 4; ++t) {
                const float pv = __expf(s[t][r] - mnew);
                s[t][r] = pv;
                rs += pv;
            }
#pragma unroll
            for (int m = 1; m <= 8; m <<= 1) rs += __shfl_xor(rs, m, 32);
            ls[r] = ls[r] * alpha + rs;
            mx[r] = mnew;
#pragma unroll
            for (int t = 0; t < 4; ++t) Oacc[t][r] *= alpha;
        }

        // P -> LDS (re-layout from C/D layout into A-fragment layout)
#pragma unroll
        for (int t = 0; t < 4; ++t)
#pragma unroll
            for (int r = 0; r < 8; ++r)
                lP[wave * 16 + hi8 + r][t * 16 + l15] = f2bf(s[t][r]);
        __syncthreads();

        // O += P @ V
#pragma unroll
        for (int kk = 0; kk < 64; kk += 32) {
            ABFrag a;
            const int prow = wave * 16 + l15;
            a.h[0] = *reinterpret_cast<const bf16x8*>(&lP[prow][kk + kh]);
            a.h[1] = *reinterpret_cast<const bf16x8*>(&lP[prow][kk + 16 + kh]);
#pragma unroll
            for (int t = 0; t < 4; ++t) {
                ABFrag bb;
                const int dcol = t * 16 + l15;
                bb.h[0] = *reinterpret_cast<const bf16x8*>(&lVt[dcol][kk + kh]);
                bb.h[1] = *reinterpret_cast<const bf16x8*>(&lVt[dcol][kk + 16 + kh]);
                Oacc[t] = WMMA_BF16(a.v, bb.v, Oacc[t]);
            }
        }
    }

    // epilogue: O /= l, write bf16 as [B*N, C] for the projection GEMM
    const int b = bh / Hh, h = bh % Hh;
#pragma unroll
    for (int r = 0; r < 8; ++r) {
        const float inv = 1.0f / ls[r];
        const int nrow = qt * 64 + wave * 16 + hi8 + r;
#pragma unroll
        for (int t = 0; t < 4; ++t) {
            const int col = h * Dd + t * 16 + l15;
            O[((size_t)(b * Nn + nrow)) * Cc + col] = f2bf(Oacc[t][r] * inv);
        }
    }
}

// ---------------------------------------------------------------------------
// Kernel 4: output projection.  attn[8192x768](bf16) @ Wproj[768x768](f32)
// + bproj -> f32 out.
// ---------------------------------------------------------------------------
__global__ __launch_bounds__(256) void proj_gemm_kernel(
    const bf16* __restrict__ A, const float* __restrict__ W,
    const float* __restrict__ bias, float* __restrict__ out)
{
    __shared__ bf16 lA[64][32];
    __shared__ bf16 lB[64][32];

    const int tid  = threadIdx.x;
    const int lane = tid & 31;
    const int wave = tid >> 5;
    const int m0   = blockIdx.y * 64;
    const int n0   = blockIdx.x * 64;
    const int wm   = (wave >> 1) * 16;
    const int wn   = (wave & 1) * 32;
    const int l15  = lane & 15;
    const int kh   = (lane >> 4) * 8;

    floatx8 acc[2];
#pragma unroll
    for (int t = 0; t < 2; ++t)
#pragma unroll
        for (int i = 0; i < 8; ++i) acc[t][i] = 0.0f;

    for (int kt = 0; kt < Cc / 32; ++kt) {
        const int k0 = kt * 32;
        {
            const int r  = tid >> 2;
            const int c0 = (tid & 3) * 8;
            const bf16* src = A + (size_t)(m0 + r) * Cc + k0 + c0;
            if (kt + 1 < Cc / 32) __builtin_prefetch(src + 32, 0, 1);
            *reinterpret_cast<bf16x8*>(&lA[r][c0]) =
                *reinterpret_cast<const bf16x8*>(src);
        }
        {
            const int n  = tid & 63;
            const int kq = (tid >> 6) * 8;
            bf16x8 pk;
#pragma unroll
            for (int i = 0; i < 8; ++i)
                pk[i] = f2bf(W[(size_t)(k0 + kq + i) * Cc + n0 + n]);
            *reinterpret_cast<bf16x8*>(&lB[n][kq]) = pk;
        }
        __syncthreads();

        ABFrag a;
        a.h[0] = *reinterpret_cast<const bf16x8*>(&lA[wm + l15][kh]);
        a.h[1] = *reinterpret_cast<const bf16x8*>(&lA[wm + l15][16 + kh]);
#pragma unroll
        for (int t = 0; t < 2; ++t) {
            ABFrag bb;
            const int n = wn + t * 16 + l15;
            bb.h[0] = *reinterpret_cast<const bf16x8*>(&lB[n][kh]);
            bb.h[1] = *reinterpret_cast<const bf16x8*>(&lB[n][16 + kh]);
            acc[t] = WMMA_BF16(a.v, bb.v, acc[t]);
        }
        __syncthreads();
    }

    const int hi8 = (lane >> 4) << 3;
#pragma unroll
    for (int t = 0; t < 2; ++t) {
        const int col = n0 + wn + t * 16 + l15;
        const float bv = bias[col];
#pragma unroll
        for (int r = 0; r < 8; ++r) {
            const int gi = m0 + wm + hi8 + r;
            out[(size_t)gi * Cc + col] = acc[t][r] + bv;
        }
    }
}

// ---------------------------------------------------------------------------
extern "C" void kernel_launch(void* const* d_in, const int* in_sizes, int n_in,
                              void* d_out, int out_size, void* d_ws, size_t ws_size,
                              hipStream_t stream) {
    (void)in_sizes; (void)n_in; (void)out_size; (void)ws_size;
    const float* x     = (const float*)d_in[0];
    const float* Wq    = (const float*)d_in[1];
    const float* bq    = (const float*)d_in[2];
    const float* Wkv   = (const float*)d_in[3];
    const float* bkv   = (const float*)d_in[4];
    const float* Wproj = (const float*)d_in[5];
    const float* bproj = (const float*)d_in[6];
    const float* scale = (const float*)d_in[7];
    float* out = (float*)d_out;

    const size_t elems = (size_t)Bb * Hh * Nn * Dd;   // 6,291,456
    bf16* qws = (bf16*)d_ws;
    bf16* kws = qws + elems;
    bf16* vws = kws + elems;
    bf16* ows = vws + elems;                          // [B*N, C] bf16

    qkv_gemm_kernel<<<dim3(NC3 / 64, BN / 64), 256, 0, stream>>>(
        x, Wq, bq, Wkv, bkv, qws, kws, vws);
    l2norm_kernel<<<(Bb * Hh * Nn) / 8, 256, 0, stream>>>(qws, scale, 1);
    l2norm_kernel<<<(Bb * Hh * Nn) / 8, 256, 0, stream>>>(kws, scale, 0);
    flash_attn_kernel<<<dim3(Nn / 64, Bb * Hh), 128, 0, stream>>>(
        qws, kws, vws, ows);
    proj_gemm_kernel<<<dim3(Cc / 64, BN / 64), 256, 0, stream>>>(
        ows, Wproj, bproj, out);
}
